// Attention_21466246545807
// MI455X (gfx1250) — matrix operand order repaired and checked by reading, not compiled, not run on hardware
//
#include <hip/hip_runtime.h>
#include <cstdint>

// ---------------------------------------------------------------------------
// MI455X (gfx1250) MHA forward: bf16 WMMA pipeline + TDM-staged GEMMs.
//   x[4,2048,1024] @ w_qkv[1024,3072] -> qkv ; per-head flash attention ;
//   attn_out @ w_out[1024,1024] + b_out -> out (fp32)
// ---------------------------------------------------------------------------

typedef __attribute__((ext_vector_type(16))) __bf16 v16bf;
typedef __attribute__((ext_vector_type(8)))  __bf16 v8bf;
typedef __attribute__((ext_vector_type(8)))  float  v8f;
typedef __attribute__((ext_vector_type(4)))  unsigned int v4u;
typedef __attribute__((ext_vector_type(8)))  int v8i;
typedef __attribute__((ext_vector_type(4)))  int v4i;

#define CAT16(lo, hi) __builtin_shufflevector(lo, hi, 0,1,2,3,4,5,6,7,8,9,10,11,12,13,14,15)

#if __has_builtin(__builtin_amdgcn_tensor_load_to_lds) && \
    __has_builtin(__builtin_amdgcn_s_wait_tensorcnt)
#define USE_TDM 1
#else
#define USE_TDM 0
#endif

// A-matrix 16x32 bf16 fragment (row-major source, leading dim `ld` elements).
// Lane<16: row=lane, K=0..7 then 16..23 ; lane>=16: row=lane-16, K=8..15 then 24..31.
__device__ __forceinline__ v16bf load_frag_a(const __bf16* base, int ld, int lane) {
  const int half = lane >> 4, r = lane & 15;
  const __bf16* p = base + (size_t)r * ld + half * 8;
  v8bf lo = *(const v8bf*)(p);
  v8bf hi = *(const v8bf*)(p + 16);
  return CAT16(lo, hi);
}

// B-matrix 32x16 bf16 fragment from Bt storage (n-major rows, contiguous K).
// Lane<16: n=lane, K=0..7 then 16..23 ; lane>=16: n=lane-16, K=8..15 then 24..31.
__device__ __forceinline__ v16bf load_frag_b(const __bf16* base, int ld, int lane) {
  const int half = lane >> 4, r = lane & 15;
  const __bf16* p = base + (size_t)r * ld + half * 8;
  v8bf lo = *(const v8bf*)(p);
  v8bf hi = *(const v8bf*)(p + 16);
  return CAT16(lo, hi);
}

__device__ __forceinline__ v8f wmma_bf16(v16bf a, v16bf b, v8f c) {
  return __builtin_amdgcn_wmma_f32_16x16x32_bf16(
      /*neg_a=*/false, a, /*neg_b=*/false, b,
      /*c_mod=*/(short)0, c, /*reuse_a=*/false, /*reuse_b=*/false);
}

#if USE_TDM
// Generic pointer to an LDS object: low 32 bits are the LDS byte offset.
__device__ __forceinline__ unsigned lds_off_u32(const void* p) {
  return (unsigned)(unsigned long long)(uintptr_t)p;
}

// TDM load of a 128-row x 32-elem bf16 tile (row stride = strideElems in the
// global tensor) into LDS at ldsOff, with HW padding of 16B after every 64B
// row => LDS row stride 80B (= LDST 40 bf16 elems).
// D# per CDNA5 ISA 8.3-8.5:
//  g0: count=1 | lds_addr | global_addr[56:0] | type=2
//  g1: data_size=2B, pad_enable, pad_interval=16DW, pad_amount=4DW,
//      tensor_dim0=32, tensor_dim1=128, tile_dim0=32, tile_dim1=128,
//      tensor_dim0_stride=strideElems
__device__ __forceinline__ void tdm_load_tile_128x32(const __bf16* gsrc,
                                                     unsigned ldsOff,
                                                     unsigned strideElems) {
  unsigned long long ga = (unsigned long long)(uintptr_t)gsrc;
  v4u g0 = { 1u,                                   // count=1 (user descriptor)
             ldsOff,                               // lds_addr [63:32]
             (unsigned)(ga & 0xFFFFFFFFu),         // global_addr lo
             (unsigned)((ga >> 32) & 0x01FFFFFFu)  // global_addr hi [120:96]
                 | 0x80000000u };                  // type=2 at [127:126]
  v8i g1 = {
      (int)((1u << 16) |        // data_size = 2 bytes
            (1u << 20) |        // pad_enable
            (3u << 22) |        // pad_interval: 16 DWORDs (64B = one row)
            (3u << 25)),        // pad_amount: 4 DWORDs (16B)
      (int)((32u & 0xFFFFu) << 16),          // tensor_dim0[15:0] @ bits 63:48
      (int)((32u >> 16) | (128u << 16)),     // tensor_dim0 hi | tensor_dim1 lo
      (int)((128u >> 16) | (32u << 16)),     // tensor_dim1 hi | tile_dim0=32
      (int)(128u),                            // tile_dim1=128, tile_dim2=0
      (int)strideElems,                       // tensor_dim0_stride lo32
      0,                                      // stride hi | dim1_stride lo (2D)
      0 };
  v4i zz4 = {0, 0, 0, 0};
  v8i zz8 = {0, 0, 0, 0, 0, 0, 0, 0};
  // 6-arg toolchain variant: (g0, g1, g2, g3, extra_group, cpol)
  __builtin_amdgcn_tensor_load_to_lds(g0, g1, zz4, zz4, zz8, /*cpol=*/0);
}
#endif

// ---------------------------------------------------------------------------
// Elementwise fp32 -> bf16
// ---------------------------------------------------------------------------
__global__ void to_bf16_kernel(const float* __restrict__ in, __bf16* __restrict__ out,
                               size_t n) {
  size_t i = (size_t)blockIdx.x * blockDim.x + threadIdx.x;
  if (i < n) out[i] = (__bf16)in[i];
}

// W[K][N] fp32 -> Wt[N][K] bf16 (coalesced read)
__global__ void transpose_to_bf16_kernel(const float* __restrict__ W,
                                         __bf16* __restrict__ Wt, int K, int N) {
  size_t i = (size_t)blockIdx.x * blockDim.x + threadIdx.x;
  size_t total = (size_t)K * N;
  if (i >= total) return;
  int n = (int)(i % N);
  int k = (int)(i / N);
  Wt[(size_t)n * K + k] = (__bf16)W[i];
}

// qkv[8192][3072] -> vT[bh=64][d=64][n=2048] (V lives at column 2048 + h*64 + d)
__global__ void build_vT_kernel(const __bf16* __restrict__ qkv, __bf16* __restrict__ vT) {
  size_t i = (size_t)blockIdx.x * blockDim.x + threadIdx.x;
  if (i >= (size_t)8192 * 1024) return;
  int colv = (int)(i & 1023);       // h*64 + d
  int m    = (int)(i >> 10);        // b*2048 + nseq
  int b = m >> 11, nseq = m & 2047;
  int h = colv >> 6, d = colv & 63;
  vT[((size_t)(b * 16 + h) * 64 + d) * 2048 + nseq] =
      qkv[(size_t)m * 3072 + 2048 + colv];
}

// ---------------------------------------------------------------------------
// Tiled bf16 WMMA GEMM: C[M][N] = A[M][K] * Bt[N][K]^T (+bias)
// Block = 256 thr = 8 waves; block tile 128x128; wave tile 32x64; K-step 32.
// TDM double-buffered staging: wave 0 issues tensor_load_to_lds for step k+1,
// s_wait_tensorcnt gates step k, one barrier per step publishes the buffer.
// ---------------------------------------------------------------------------
template <typename OutT, bool HAS_BIAS>
__global__ void __launch_bounds__(256)
gemm_bf16_wmma(const __bf16* __restrict__ A, const __bf16* __restrict__ Bt,
               OutT* __restrict__ C, const float* __restrict__ bias,
               int M, int N, int K) {
  constexpr int BM = 128, BN = 128, KT = 32, LDST = 40;  // 80B LDS rows, 16B aligned
  __shared__ __bf16 As[2][BM * LDST];
  __shared__ __bf16 Bs[2][BN * LDST];

  const int t = threadIdx.x;
  const int lane = t & 31, wid = t >> 5;
  const int wm = (wid >> 1) * 32;   // wave row offset in block tile
  const int wn = (wid & 1) * 64;    // wave col offset
  const int m0 = blockIdx.y * BM, n0 = blockIdx.x * BN;

  v8f acc[2][4] = {};

#if USE_TDM
  // Escape the LDS base pointers through asm so the compiler cannot conclude
  // the arrays are never written (TDM writes them behind its back).
  __bf16* asp = &As[0][0];
  __bf16* bsp = &Bs[0][0];
  asm volatile("" : "+v"(asp), "+v"(bsp) :: "memory");

  const bool issuer = (wid == 0);
  const unsigned asOff = lds_off_u32(asp);
  const unsigned bsOff = lds_off_u32(bsp);
  constexpr unsigned BUFB = (unsigned)(BM * LDST * sizeof(__bf16));  // 10240 B

  if (issuer) {
    tdm_load_tile_128x32(A + (size_t)m0 * K, asOff, (unsigned)K);
    tdm_load_tile_128x32(Bt + (size_t)n0 * K, bsOff, (unsigned)K);
  }

  const int steps = K / KT;
  for (int kt = 0; kt < steps; ++kt) {
    const int cur = kt & 1;
    if (issuer) __builtin_amdgcn_s_wait_tensorcnt(0);
    __syncthreads();  // publish buf[cur]; all waves done with buf[cur^1]

    const int kn = (kt + 1) * KT;
    if (issuer && kn < K) {
      tdm_load_tile_128x32(A + (size_t)m0 * K + kn, asOff + (cur ^ 1) * BUFB,
                           (unsigned)K);
      tdm_load_tile_128x32(Bt + (size_t)n0 * K + kn, bsOff + (cur ^ 1) * BUFB,
                           (unsigned)K);
    }

    const __bf16* Ap = asp + cur * (BM * LDST);
    const __bf16* Bp = bsp + cur * (BN * LDST);
    v16bf af[2], bfr[4];
    for (int fm = 0; fm < 2; ++fm)
      af[fm] = load_frag_a(Ap + (wm + fm * 16) * LDST, LDST, lane);
    for (int fn = 0; fn < 4; ++fn)
      bfr[fn] = load_frag_b(Bp + (wn + fn * 16) * LDST, LDST, lane);
    for (int fm = 0; fm < 2; ++fm)
      for (int fn = 0; fn < 4; ++fn)
        acc[fm][fn] = wmma_bf16(af[fm], bfr[fn], acc[fm][fn]);
  }
#else
  for (int k0 = 0; k0 < K; k0 += KT) {
    for (int i = 0; i < 2; ++i) {
      int c = t + i * 256;
      int row = c >> 2, col = (c & 3) * 8;
      *(v8bf*)(&As[0][0] + row * LDST + col) =
          *(const v8bf*)(A + (size_t)(m0 + row) * K + k0 + col);
      *(v8bf*)(&Bs[0][0] + row * LDST + col) =
          *(const v8bf*)(Bt + (size_t)(n0 + row) * K + k0 + col);
    }
    __syncthreads();
    v16bf af[2], bfr[4];
    for (int fm = 0; fm < 2; ++fm)
      af[fm] = load_frag_a(&As[0][0] + (wm + fm * 16) * LDST, LDST, lane);
    for (int fn = 0; fn < 4; ++fn)
      bfr[fn] = load_frag_b(&Bs[0][0] + (wn + fn * 16) * LDST, LDST, lane);
    for (int fm = 0; fm < 2; ++fm)
      for (int fn = 0; fn < 4; ++fn)
        acc[fm][fn] = wmma_bf16(af[fm], bfr[fn], acc[fm][fn]);
    __syncthreads();
  }
#endif

  // C/D layout: VGPR j, lanes 0-15 -> M=j ; lanes 16-31 -> M=j+8 ; N = lane&15.
  const int rsel = (lane >> 4) * 8, nc = lane & 15;
  for (int fm = 0; fm < 2; ++fm)
    for (int fn = 0; fn < 4; ++fn)
      for (int j = 0; j < 8; ++j) {
        int m = m0 + wm + fm * 16 + rsel + j;
        int n = n0 + wn + fn * 16 + nc;
        float v = acc[fm][fn][j];
        if (HAS_BIAS) v += bias[n];
        C[(size_t)m * N + n] = (OutT)v;
      }
}

// ---------------------------------------------------------------------------
// Flash attention: one wave per 16-row Q tile. S = Q K^T (WMMA), online
// softmax (shfl_xor over 16-lane halves), P staged through per-wave LDS to
// re-layout C->A fragments, O += P V (WMMA) with pre-transposed V^T.
// ---------------------------------------------------------------------------
__global__ void __launch_bounds__(256)
attn_kernel(const __bf16* __restrict__ qkv, const __bf16* __restrict__ vT,
            __bf16* __restrict__ outp) {
  constexpr int SEQ = 2048, LD = 3072;
  __shared__ __bf16 Ps[8][16 * 32];   // per-wave P scratch (16 rows x 32 keys)

  const int lane = threadIdx.x & 31, wid = threadIdx.x >> 5;
  const int bh = blockIdx.y;
  const int b = bh >> 4, h = bh & 15;
  const int qrow = b * SEQ + blockIdx.x * 128 + wid * 16;  // global row in [8192]
  const int qcol = h * 64;
  const float scale = 0.03125f;  // 1024^-0.5 (reference scales by full dim)

  const __bf16* Qbase = qkv + (size_t)qrow * LD + qcol;
  const __bf16* Kbase = qkv + (size_t)(b * SEQ) * LD + 1024 + qcol;
  const __bf16* Vbase = vT + (size_t)bh * 64 * SEQ;  // rows = d (64), ld = SEQ

  v16bf qf[2];
  qf[0] = load_frag_a(Qbase, LD, lane);        // d = 0..31
  qf[1] = load_frag_a(Qbase + 32, LD, lane);   // d = 32..63

  v8f o[4] = {};
  float mrun[8], lrun[8];
  for (int j = 0; j < 8; ++j) { mrun[j] = -1e30f; lrun[j] = 0.0f; }

  for (int j0 = 0; j0 < SEQ; j0 += 32) {
    // S tiles for keys [j0, j0+16) and [j0+16, j0+32)
    v8f s0 = {}, s1 = {};
    for (int c = 0; c < 2; ++c) {
      v16bf k0f = load_frag_b(Kbase + (size_t)j0 * LD + c * 32, LD, lane);
      v16bf k1f = load_frag_b(Kbase + (size_t)(j0 + 16) * LD + c * 32, LD, lane);
      s0 = wmma_bf16(qf[c], k0f, s0);
      s1 = wmma_bf16(qf[c], k1f, s1);
    }

    float p0[8], p1[8], corr[8];
    for (int j = 0; j < 8; ++j) {
      float a  = s0[j] * scale;
      float bb = s1[j] * scale;
      float mx = fmaxf(a, bb);
      for (int off = 8; off > 0; off >>= 1)
        mx = fmaxf(mx, __shfl_xor(mx, off, 32));     // row-max within 16-lane half
      float mnew = fmaxf(mrun[j], mx);
      float cr = __expf(mrun[j] - mnew);
      float e0 = __expf(a - mnew);
      float e1 = __expf(bb - mnew);
      float sm = e0 + e1;
      for (int off = 8; off > 0; off >>= 1)
        sm += __shfl_xor(sm, off, 32);               // row-sum within 16-lane half
      lrun[j] = lrun[j] * cr + sm;
      mrun[j] = mnew;
      corr[j] = cr;
      p0[j] = e0;
      p1[j] = e1;
    }

    for (int fn = 0; fn < 4; ++fn)
      for (int j = 0; j < 8; ++j) o[fn][j] *= corr[j];

    // C layout -> row-major P in LDS (bf16), then reload as A fragment.
    {
      const int rbase = (lane >> 4) * 8, cc = lane & 15;
      for (int j = 0; j < 8; ++j) {
        Ps[wid][(rbase + j) * 32 + cc]      = (__bf16)p0[j];
        Ps[wid][(rbase + j) * 32 + 16 + cc] = (__bf16)p1[j];
      }
    }
    v16bf pa = load_frag_a(&Ps[wid][0], 32, lane);  // same-wave DS ops are in-order

    // O(16x64) += P(16x32) * V(32x64): V^T rows are d-columns, contiguous keys.
    for (int fn = 0; fn < 4; ++fn) {
      v16bf vf = load_frag_b(Vbase + (size_t)(fn * 16) * SEQ + j0, SEQ, lane);
      o[fn] = wmma_bf16(pa, vf, o[fn]);
    }
  }

  const int rbase = (lane >> 4) * 8, cc = lane & 15;
  for (int fn = 0; fn < 4; ++fn)
    for (int j = 0; j < 8; ++j) {
      float v = o[fn][j] / lrun[j];
      outp[(size_t)(qrow + rbase + j) * 1024 + qcol + fn * 16 + cc] = (__bf16)v;
    }
}

// ---------------------------------------------------------------------------
// Host-side launch
// ---------------------------------------------------------------------------
extern "C" void kernel_launch(void* const* d_in, const int* in_sizes, int n_in,
                              void* d_out, int out_size, void* d_ws, size_t ws_size,
                              hipStream_t stream) {
  (void)in_sizes; (void)n_in; (void)out_size; (void)ws_size;
  const float* x     = (const float*)d_in[0];  // [4,2048,1024]
  const float* w_qkv = (const float*)d_in[1];  // [1024,3072]
  const float* w_out = (const float*)d_in[2];  // [1024,1024]
  const float* b_out = (const float*)d_in[3];  // [1024]
  float* out = (float*)d_out;                  // [4,2048,1024]

  constexpr size_t M = 8192;   // 4*2048
  constexpr size_t D = 1024;
  constexpr size_t N3 = 3072;

  __bf16* xb    = (__bf16*)d_ws;                  // M*D
  __bf16* wqkvT = xb    + M * D;                  // N3*D   (transposed)
  __bf16* woutT = wqkvT + N3 * D;                 // D*D    (transposed)
  __bf16* qkv   = woutT + D * D;                  // M*N3
  __bf16* vT    = qkv   + M * N3;                 // 64*64*2048
  __bf16* aout  = vT    + (size_t)64 * 64 * 2048; // M*D

  to_bf16_kernel<<<dim3((unsigned)((M * D) / 256)), dim3(256), 0, stream>>>(x, xb, M * D);
  transpose_to_bf16_kernel<<<dim3((unsigned)((D * N3) / 256)), dim3(256), 0, stream>>>(
      w_qkv, wqkvT, (int)D, (int)N3);
  transpose_to_bf16_kernel<<<dim3((unsigned)((D * D) / 256)), dim3(256), 0, stream>>>(
      w_out, woutT, (int)D, (int)D);

  // QKV projection: [8192,1024] x [1024,3072] -> bf16 qkv
  gemm_bf16_wmma<__bf16, false><<<dim3((unsigned)(N3 / 128), (unsigned)(M / 128)),
                                  dim3(256), 0, stream>>>(
      xb, wqkvT, qkv, nullptr, (int)M, (int)N3, (int)D);

  build_vT_kernel<<<dim3((unsigned)((M * D) / 256)), dim3(256), 0, stream>>>(qkv, vT);

  // Flash attention: 16 q-tiles of 128 rows per (b,h); 64 (b,h) pairs.
  attn_kernel<<<dim3(16, 64), dim3(256), 0, stream>>>(qkv, vT, aout);

  // Output projection + bias: [8192,1024] x [1024,1024] -> fp32 out
  gemm_bf16_wmma<float, true><<<dim3((unsigned)(D / 128), (unsigned)(M / 128)),
                                dim3(256), 0, stream>>>(
      aout, woutT, out, b_out, (int)M, (int)D, (int)D);
}
